// Position_Attention_7791070675413
// MI455X (gfx1250) — compile-verified, matrix operand order checked by reading, not run on hardware
//
#include <hip/hip_runtime.h>
#include <hip/hip_bf16.h>

typedef __attribute__((ext_vector_type(16))) __bf16 v16bf;
typedef __attribute__((ext_vector_type(8)))  __bf16 v8bf;
typedef __attribute__((ext_vector_type(8)))  float  v8f;

// ---------------------------------------------------------------- utilities

__device__ __forceinline__ float my_tanh(float x) {
#if __has_builtin(__builtin_amdgcn_tanhf)
  return __builtin_amdgcn_tanhf(x);
#elif __has_builtin(__builtin_amdgcn_tanh_f32)
  return __builtin_amdgcn_tanh_f32(x);
#else
  return tanhf(x);
#endif
}

union FragU { v16bf v; v8bf h[2]; };

// ---------------------------------------------------------------- prep
// grid (H/256, B*T), block 256. No integer division anywhere.
__global__ __launch_bounds__(256) void prep_hs_kernel(const float* __restrict__ hs,
                                                      __bf16* __restrict__ hs_bf,
                                                      __bf16* __restrict__ hsT,
                                                      float* __restrict__ outAll) {
  constexpr int H = 1536, T = 256;
  const int m = blockIdx.y;                       // b*T + s
  const int h = blockIdx.x * 256 + threadIdx.x;
  const int b = m >> 8, s = m & 255;
  const size_t i = (size_t)m * H + h;
  const float x = hs[i];
  hs_bf[i] = (__bf16)x;
  hsT[((size_t)b * H + h) * T + s] = (__bf16)x;   // transposed copy for c-GEMM B
  outAll[(size_t)m * (2 * H) + h] = x;            // all_ut[..., 0:H] = hs
}

__global__ __launch_bounds__(256) void convert_bf16_kernel(const float* __restrict__ src,
                                                           __bf16* __restrict__ dst,
                                                           long long n) {
  for (long long i = (long long)blockIdx.x * blockDim.x + threadIdx.x; i < n;
       i += (long long)gridDim.x * blockDim.x)
    dst[i] = (__bf16)src[i];
}

// p[b,o] = sum_k hs[b,qid,k] * W_p[o,k]  (tiny, fp32)
__global__ __launch_bounds__(256) void compute_p_kernel(const float* __restrict__ hs,
                                                        const float* __restrict__ W_p,
                                                        const int* __restrict__ qid,
                                                        float* __restrict__ P) {
  constexpr int T = 256, H = 1536;
  const int b = blockIdx.y;
  const int o = blockIdx.x * 256 + threadIdx.x;
  const int q = qid[0];
  const float* hrow = hs + ((size_t)(b * T + q)) * H;
  const float* wrow = W_p + (size_t)o * H;
  float acc = 0.f;
  for (int k = 0; k < H; ++k) acc = fmaf(hrow[k], wrow[k], acc);
  P[b * H + o] = acc;
}

// ---------------------------------------------------------------- WMMA GEMM
// C[m,n] = sum_k A[m,k] * Bt[n,k]   (Bt row-major N x K == B^T)
// One wave per workgroup, 64(M) x 64(N) tile, K stepped by 32:
// 16 v_wmma per K-step against 16 b128 fragment loads (1.0 loads/wmma).
__global__ __launch_bounds__(32) void wmma_gemm_bf16_kernel(
    const __bf16* __restrict__ A, long long aBatch, int lda,
    const __bf16* __restrict__ Bt, long long bBatch, int ldb,
    float* __restrict__ C, long long cBatch, int ldc,
    int K,
    const float* __restrict__ addP, int addLd, int rowsPerBatch) {
  const int lane = threadIdx.x & 31;
  const int n0 = blockIdx.x * 64;
  const int m0 = blockIdx.y * 64;
  const int bz = blockIdx.z;
  const __bf16* Ab = A + (size_t)bz * aBatch;
  const __bf16* Bb = Bt + (size_t)bz * bBatch;
  float* Cb = C + (size_t)bz * cBatch;

  const int r15 = lane & 15;
  const int hi = (lane >> 4) & 1;
  const int koff = hi ? 8 : 0;   // 16-bit A-layout: lanes 16-31 start at K+8 / K+24

  // Per-lane row pointers, fixed for the whole K loop.
  const __bf16* pa[4];
  const __bf16* pb[4];
#pragma unroll
  for (int i = 0; i < 4; ++i) pa[i] = Ab + (size_t)(m0 + 16 * i + r15) * lda + koff;
#pragma unroll
  for (int j = 0; j < 4; ++j) pb[j] = Bb + (size_t)(n0 + 16 * j + r15) * ldb + koff;

  v8f acc[4][4];
#pragma unroll
  for (int i = 0; i < 4; ++i)
#pragma unroll
    for (int j = 0; j < 4; ++j) acc[i][j] = (v8f)0.0f;

  for (int k0 = 0; k0 < K; k0 += 32) {
    v16bf aF[4], bF[4];
#pragma unroll
    for (int i = 0; i < 4; ++i) {
      FragU f;
      f.h[0] = *(const v8bf*)(pa[i] + k0);
      f.h[1] = *(const v8bf*)(pa[i] + k0 + 16);
      aF[i] = f.v;
    }
#pragma unroll
    for (int j = 0; j < 4; ++j) {
      FragU f;
      f.h[0] = *(const v8bf*)(pb[j] + k0);
      f.h[1] = *(const v8bf*)(pb[j] + k0 + 16);
      bF[j] = f.v;
    }
    if (k0 + 256 < K) {            // long-lookahead prefetch (global_prefetch_b8)
      __builtin_prefetch(pa[0] + k0 + 256, 0, 3);
      __builtin_prefetch(pb[0] + k0 + 256, 0, 3);
    }
#pragma unroll
    for (int i = 0; i < 4; ++i)
#pragma unroll
      for (int j = 0; j < 4; ++j)
        acc[i][j] = __builtin_amdgcn_wmma_f32_16x16x32_bf16(
            false, aF[i], false, bF[j], (short)0, acc[i][j], false, false);
  }

  // Epilogue: D layout = 8 VGPRs; lane L -> n = (L&15), m = 8*(L>>4) + r
  const float* addRow = nullptr;
  if (addP) addRow = addP + (size_t)(m0 / rowsPerBatch) * addLd;
  const int mhalf = hi * 8;
#pragma unroll
  for (int i = 0; i < 4; ++i) {
#pragma unroll
    for (int j = 0; j < 4; ++j) {
      const int nn = n0 + 16 * j + r15;
      const float av = addRow ? addRow[nn] : 0.0f;
#pragma unroll
      for (int r = 0; r < 8; ++r)
        Cb[(size_t)(m0 + 16 * i + mhalf + r) * ldc + nn] = acc[i][j][r] + av;
    }
  }
}

// ---------------------------------------------------------------- scores+softmax
// One workgroup (8 wave32s) per (b,t). Each wave reduces score rows s.
__global__ __launch_bounds__(256) void scores_softmax_kernel(
    const float* __restrict__ WHX, const float* __restrict__ TAIL,
    const float* __restrict__ v, float* __restrict__ Af,
    __bf16* __restrict__ Abf) {
  constexpr int T = 256, S = 256, H = 1536;
  __shared__ __align__(16) float tailLDS[H];
  __shared__ __align__(16) float vLDS[H];
  __shared__ float sc[S];
  __shared__ float red[256];
  const int t = blockIdx.x, b = blockIdx.y;
  const int tid = threadIdx.x;
  const float* tailRow = TAIL + ((size_t)(b * T + t)) * H;
  for (int i = tid; i < H; i += 256) { tailLDS[i] = tailRow[i]; vLDS[i] = v[i]; }
  __syncthreads();

  const int lane = tid & 31, wave = tid >> 5;
  constexpr int Q = H >> 2;  // float4 chunks
  for (int s = wave; s < S; s += 8) {
    const float4* wr4 = (const float4*)(WHX + ((size_t)(b * S + s)) * H);
    float acc = 0.f;
    for (int q = lane; q < Q; q += 32) {
      float4 wv = wr4[q];
      float4 tv = *(const float4*)&tailLDS[q * 4];
      float4 vv = *(const float4*)&vLDS[q * 4];
      acc += my_tanh(wv.x + tv.x) * vv.x;
      acc += my_tanh(wv.y + tv.y) * vv.y;
      acc += my_tanh(wv.z + tv.z) * vv.z;
      acc += my_tanh(wv.w + tv.w) * vv.w;
    }
#pragma unroll
    for (int off = 16; off; off >>= 1) acc += __shfl_xor(acc, off, 32);
    if (lane == 0) sc[s] = acc;
  }
  __syncthreads();

  const float x = sc[tid];
  red[tid] = x; __syncthreads();
  for (int st = 128; st; st >>= 1) {
    if (tid < st) red[tid] = fmaxf(red[tid], red[tid + st]);
    __syncthreads();
  }
  const float mx = red[0]; __syncthreads();
  const float e = __expf(x - mx);
  red[tid] = e; __syncthreads();
  for (int st = 128; st; st >>= 1) {
    if (tid < st) red[tid] += red[tid + st];
    __syncthreads();
  }
  const float a = e / red[0];
  const size_t idx = ((size_t)(b * T + t)) * S + tid;
  Af[idx] = a;
  Abf[idx] = (__bf16)a;
}

// ---------------------------------------------------------------- pool + normalize
__global__ __launch_bounds__(256) void pool_norm_kernel(const float* __restrict__ Af,
                                                        float* __restrict__ out0) {
  constexpr int T = 256, S = 256;
  const int b = blockIdx.x;
  const int s = threadIdx.x;
  float acc = 0.f;
  for (int t = 0; t < T; ++t) acc += Af[((size_t)(b * T + t)) * S + s];
  acc *= (1.0f / (float)T);
  __shared__ float red[256];
  red[s] = acc; __syncthreads();
  for (int st = 128; st; st >>= 1) {
    if (s < st) red[s] += red[s + st];
    __syncthreads();
  }
  float nv = acc / red[0];
  if (!(nv == nv)) nv = 0.f;  // NaN guard per reference
  out0[b * S + s] = nv;
}

// ---------------------------------------------------------------- launcher

extern "C" void kernel_launch(void* const* d_in, const int* in_sizes, int n_in,
                              void* d_out, int out_size, void* d_ws, size_t ws_size,
                              hipStream_t stream) {
  constexpr int B = 2, T = 256, S = 256, H = 1536;
  const float* hs  = (const float*)d_in[0];
  const float* W_H = (const float*)d_in[1];
  const float* W_p = (const float*)d_in[2];
  const float* W_h = (const float*)d_in[3];
  const float* v   = (const float*)d_in[4];
  const int*   qid = (const int*)d_in[5];

  float* out0   = (float*)d_out;                 // normalized_at: B*S
  float* outAll = out0 + (size_t)B * S;          // all_ut: B*T*2H

  char* w = (char*)d_ws;
  auto take = [&](size_t bytes) -> char* {
    char* p = w;
    w += (bytes + 255) & ~(size_t)255;
    return p;
  };
  __bf16* hs_bf = (__bf16*)take((size_t)B * T * H * 2);
  __bf16* WHbf  = (__bf16*)take((size_t)H * H * 2);
  __bf16* Whbf  = (__bf16*)take((size_t)H * H * 2);
  __bf16* hsT   = (__bf16*)take((size_t)B * H * T * 2);
  float*  P     = (float*)take((size_t)B * H * 4);
  float*  WHX   = (float*)take((size_t)B * S * H * 4);
  float*  TAIL  = (float*)take((size_t)B * T * H * 4);
  float*  Af    = (float*)take((size_t)B * T * S * 4);
  __bf16* Abf   = (__bf16*)take((size_t)B * T * S * 2);

  // Stage 0: conversions / transposes / hs -> all_ut[:, :H]
  prep_hs_kernel<<<dim3(H / 256, B * T), 256, 0, stream>>>(hs, hs_bf, hsT, outAll);
  convert_bf16_kernel<<<4096, 256, 0, stream>>>(W_H, WHbf, (long long)H * H);
  convert_bf16_kernel<<<4096, 256, 0, stream>>>(W_h, Whbf, (long long)H * H);
  compute_p_kernel<<<dim3(H / 256, B), 256, 0, stream>>>(hs, W_p, qid, P);

  // Stage 1: WHX = hs @ W_H^T ; TAIL = hs @ W_h^T + p (fused epilogue)
  wmma_gemm_bf16_kernel<<<dim3(H / 64, (B * T) / 64, 1), 32, 0, stream>>>(
      hs_bf, 0, H, WHbf, 0, H, WHX, 0, H, H, nullptr, 0, 256);
  wmma_gemm_bf16_kernel<<<dim3(H / 64, (B * T) / 64, 1), 32, 0, stream>>>(
      hs_bf, 0, H, Whbf, 0, H, TAIL, 0, H, H, P, H, T);

  // Stage 2: scores + softmax (fp32 VALU, tanh-dominated)
  scores_softmax_kernel<<<dim3(T, B), 256, 0, stream>>>(WHX, TAIL, v, Af, Abf);

  // Stage 3: c = a @ hs, written into all_ut[:, H:2H] (batched over grid.z)
  wmma_gemm_bf16_kernel<<<dim3(H / 64, T / 64, B), 32, 0, stream>>>(
      Abf, (long long)T * S, S, hsT, (long long)H * T, T,
      outAll + H, (long long)T * 2 * H, 2 * H, S, nullptr, 0, 256);

  // Stage 4: at_pool mean over t, normalize, NaN guard
  pool_norm_kernel<<<dim3(B), 256, 0, stream>>>(Af, out0);
}
